// Tier1MoE_24206435680282
// MI455X (gfx1250) — compile-verified, matrix operand order checked by reading, not run on hardware
//
#include <hip/hip_runtime.h>

// ---------------- problem constants ----------------
constexpr int B_  = 64;
constexpr int N_  = 1024;
constexpr int D_  = 256;
constexpr int E_  = 8;
constexpr int NF_ = 4;
constexpr int G4_ = 4 * D_;     // 1024 gate columns
constexpr int H_  = 4;
constexpr int DH_ = D_ / H_;    // 64

// ---------------- WMMA plumbing (gfx1250, wave32) ----------------
typedef __attribute__((ext_vector_type(16))) __bf16 v16bf;
typedef __attribute__((ext_vector_type(8)))  float  v8f;

union V16U { v16bf v; uint4 q[2]; };

// Load a 16x32 bf16 operand tile from a row-major matrix (leading dim `ld`).
// Layout per CDNA5 ISA 7.12.2 (16-bit A/B): lanes 0-15 hold K {k0..k0+7, k0+16..k0+23},
// lanes 16-31 hold K {k0+8..k0+15, k0+24..k0+31}; two contiguous 16B loads per lane.
static __device__ __forceinline__ v16bf ld_tile16(const __bf16* base, int ld,
                                                  int row0, int k0, int lane) {
  const int r  = row0 + (lane & 15);
  const int kk = k0 + ((lane & 16) ? 8 : 0);
  const __bf16* p = base + (long)r * ld + kk;
  V16U u;
  u.q[0] = *(const uint4*)(p);
  u.q[1] = *(const uint4*)(p + 16);
  return u.v;
}

static __device__ __forceinline__ v8f wmma_bf16(v16bf a, v16bf b, v8f c) {
  // D = A(16x32) * B(32x16) + C(16x16 f32)
  return __builtin_amdgcn_wmma_f32_16x16x32_bf16(false, a, false, b, (short)0, c,
                                                 false, false);
}

static __device__ __forceinline__ float sigm(float x) {
  return 1.0f / (1.0f + __expf(-x));
}

// Compiler memory barrier: stops LICM/CSE from hoisting loop-invariant WMMA
// operand tiles across iterations (spill-to-scratch prevention).
static __device__ __forceinline__ void antihoist() {
  asm volatile("" ::: "memory");
}

// ---------------- async Global <-> LDS DMA (gfx1250 ASYNCcnt path) ----------------
// Generic flat pointers to LDS carry the wave-relative LDS byte address in the
// low 32 bits (ISA 10.2 aperture mapping), which is what VDST/VSRC expect.
static __device__ __forceinline__ unsigned lds_addr_of(const void* p) {
  return (unsigned)(size_t)p;
}
static __device__ __forceinline__ void async_load_b128(unsigned lds, const void* g) {
  asm volatile("global_load_async_to_lds_b128 %0, %1, off" :: "v"(lds), "v"(g) : "memory");
}
static __device__ __forceinline__ void async_store_b128(const void* g, unsigned lds) {
  asm volatile("global_store_async_from_lds_b128 %0, %1, off" :: "v"(g), "v"(lds) : "memory");
}
static __device__ __forceinline__ void wait_async0() {
  asm volatile("s_wait_asynccnt 0x0" ::: "memory");
}

// ---------------- small utility kernels ----------------
__global__ void cvt_kernel(const float* __restrict__ s, __bf16* __restrict__ d, long n) {
  long stride = (long)gridDim.x * blockDim.x;
  for (long i = (long)blockIdx.x * blockDim.x + threadIdx.x; i < n; i += stride)
    d[i] = (__bf16)s[i];
}

__global__ void zmean_kernel(const float* __restrict__ Z, __bf16* __restrict__ zmb) {
  int b = blockIdx.x, d = threadIdx.x;
  float s = 0.f;
  for (int t = 0; t < N_; ++t) s += Z[((long)b * N_ + t) * D_ + d];
  zmb[b * D_ + d] = (__bf16)(s * (1.0f / (float)N_));
}

// ---------------- fused LSTM encoder (1024 steps) + decoder (4 steps) ----------------
// One 512-thread block per expert. h double-buffered in LDS (bf16); c state in LDS
// (f32, unique owner thread per element); per step the block performs the full
// g = [x_t, h] @ [Wih, Whh]^T + b as 4096 v_wmma_f32_16x16x32_bf16 (256 per wave).
// x_t staged and h sequence written back via async Global<->LDS DMA (ASYNCcnt).
__global__ void lstm_scan_kernel(
    const __bf16* __restrict__ Zbf,  const float* __restrict__ c_h,
    const __bf16* __restrict__ WihE, const __bf16* __restrict__ WhhE, const float* __restrict__ bE,
    const __bf16* __restrict__ WihD, const __bf16* __restrict__ WhhD, const float* __restrict__ bD,
    const float* __restrict__ Wh0,   const float* __restrict__ bh0,
    const float* __restrict__ Wc0,   const float* __restrict__ bc0,
    __bf16* __restrict__ enc, __bf16* __restrict__ dec)
{
  const int e    = blockIdx.x;
  const int tid  = threadIdx.x;           // 512 threads = 16 waves
  const int wave = tid >> 5;
  const int lane = tid & 31;
  const int mlo  = (lane & 16) ? 8 : 0;   // C-layout row offset
  const int nlo  = lane & 15;             // C-layout column

  extern __shared__ unsigned char sm[];
  __bf16* x_sh  = (__bf16*)sm;            // 64 x 256 bf16   (32 KB)
  __bf16* hbuf0 = x_sh  + B_ * D_;        //                 (32 KB)
  __bf16* hbuf1 = hbuf0 + B_ * D_;        //                 (32 KB)
  float*  c_sh  = (float*)(hbuf1 + B_ * D_); // 64 x 256 f32 (64 KB)
  __bf16* h_cur = hbuf0;
  __bf16* h_nxt = hbuf1;

  const long eW = (long)e * G4_ * D_;
  const int  mrow = tid >> 3;             // staging row (b index)
  const int  mcol = (tid & 7) * 32;       // staging column base (bf16 elems)

  // ---- init h0 = c_h@Wh0^T + bh0 ; c0 = c_h@Wc0^T + bc0 (K=32, VALU) ----
  #pragma unroll
  for (int p = 0; p < 4; ++p) {
    const int pr = wave * 4 + p;
    const int bt = pr >> 4, jt = pr & 15;
    #pragma unroll
    for (int r = 0; r < 8; ++r) {
      const int m = bt * 16 + r + mlo;
      const int n = jt * 16 + nlo;
      float hv = bh0[e * D_ + n], cv = bc0[e * D_ + n];
      for (int s = 0; s < 32; ++s) {
        const float ch = c_h[m * 32 + s];
        hv += ch * Wh0[((long)e * D_ + n) * 32 + s];
        cv += ch * Wc0[((long)e * D_ + n) * 32 + s];
      }
      c_sh[m * D_ + n]  = cv;
      h_cur[m * D_ + n] = (__bf16)hv;
    }
  }
  __syncthreads();

  for (int phase = 0; phase < 2; ++phase) {
    const __bf16* Wx   = (phase == 0) ? (WihE + eW) : (WihD + eW);
    const __bf16* Wh   = (phase == 0) ? (WhhE + eW) : (WhhD + eW);
    const float*  bb   = (phase == 0) ? (bE + (long)e * G4_) : (bD + (long)e * G4_);
    __bf16*       outp = (phase == 0) ? (enc + (long)e * N_ * B_ * D_)
                                      : (dec + (long)e * NF_ * B_ * D_);
    const int steps    = (phase == 0) ? N_ : NF_;

    // hoist biases (lane/pair dependent, t-invariant SSA values; added in the
    // pointwise phase so accumulators can zero-init -> inline SRC2=0, no spills)
    float biasr[16];
    #pragma unroll
    for (int p = 0; p < 4; ++p) {
      const int jt = (wave * 4 + p) & 15;
      #pragma unroll
      for (int g = 0; g < 4; ++g) biasr[p * 4 + g] = bb[g * D_ + jt * 16 + nlo];
    }

    if (phase == 1) {
      // decoder input = enc[-1] broadcast: freeze current h into x_sh
      const uint4* s4 = (const uint4*)(h_cur + mrow * D_ + mcol);
      uint4*       d4 = (uint4*)(x_sh + mrow * D_ + mcol);
      d4[0] = s4[0]; d4[1] = s4[1]; d4[2] = s4[2]; d4[3] = s4[3];
    }

    for (int t = 0; t < steps; ++t) {
      if (phase == 0) {
        // stage x_t = Z[:, t, :] into LDS via async DMA (no VGPR round trip)
        const __bf16* g = Zbf + ((long)mrow * N_ + t) * D_ + mcol;
        const unsigned l = lds_addr_of(x_sh + mrow * D_ + mcol);
        async_load_b128(l,      g);
        async_load_b128(l + 16, g + 8);
        async_load_b128(l + 32, g + 16);
        async_load_b128(l + 48, g + 24);
        // pull next step's x into near cache while this step computes
        if (t + 1 < steps)
          __builtin_prefetch(Zbf + ((long)mrow * N_ + (t + 1)) * D_ + mcol, 0, 3);
      }
      wait_async0();     // x_t staged; also retires h-sequence stores from t-1/t-2
      __syncthreads();   // x_sh staged, h_cur stable

      #pragma unroll
      for (int p = 0; p < 4; ++p) {
        antihoist();     // keep tile loads inside the loop; no cross-pair CSE
        const int pr = wave * 4 + p;
        const int bt = pr >> 4, jt = pr & 15;

        v8f a0 = {}, a1 = {}, a2 = {}, a3 = {};
        // x contribution (K = 256)
        #pragma unroll
        for (int ks = 0; ks < 8; ++ks) {
          const v16bf av = ld_tile16(x_sh, D_, bt * 16, ks * 32, lane);
          a0 = wmma_bf16(av, ld_tile16(Wx, D_, 0 * D_ + jt * 16, ks * 32, lane), a0);
          a1 = wmma_bf16(av, ld_tile16(Wx, D_, 1 * D_ + jt * 16, ks * 32, lane), a1);
          a2 = wmma_bf16(av, ld_tile16(Wx, D_, 2 * D_ + jt * 16, ks * 32, lane), a2);
          a3 = wmma_bf16(av, ld_tile16(Wx, D_, 3 * D_ + jt * 16, ks * 32, lane), a3);
        }
        // h contribution (K = 256)
        #pragma unroll
        for (int ks = 0; ks < 8; ++ks) {
          const v16bf av = ld_tile16(h_cur, D_, bt * 16, ks * 32, lane);
          a0 = wmma_bf16(av, ld_tile16(Wh, D_, 0 * D_ + jt * 16, ks * 32, lane), a0);
          a1 = wmma_bf16(av, ld_tile16(Wh, D_, 1 * D_ + jt * 16, ks * 32, lane), a1);
          a2 = wmma_bf16(av, ld_tile16(Wh, D_, 2 * D_ + jt * 16, ks * 32, lane), a2);
          a3 = wmma_bf16(av, ld_tile16(Wh, D_, 3 * D_ + jt * 16, ks * 32, lane), a3);
        }
        // pointwise LSTM cell (C layout: element r,lane -> (m,n)); bias added here
        #pragma unroll
        for (int r = 0; r < 8; ++r) {
          const float iv = a0[r] + biasr[p * 4 + 0];
          const float fv = a1[r] + biasr[p * 4 + 1];
          const float gv = a2[r] + biasr[p * 4 + 2];
          const float ov = a3[r] + biasr[p * 4 + 3];
          const int m = bt * 16 + r + mlo;
          const int n = jt * 16 + nlo;
          const float cp = c_sh[m * D_ + n];
          const float c2 = sigm(fv) * cp + sigm(iv) * tanhf(gv);
          const float h2 = sigm(ov) * tanhf(c2);
          c_sh[m * D_ + n]  = c2;            // unique owner: no race
          h_nxt[m * D_ + n] = (__bf16)h2;    // double-buffered: no WAR on h_cur
        }
      }
      __syncthreads();   // h_nxt complete; all reads of x_sh/h_cur done

      // bulk-write this step's h to the sequence output: LDS -> global async DMA
      {
        __bf16* gdst = outp + ((long)t * B_ + mrow) * D_ + mcol;
        const unsigned l = lds_addr_of(h_nxt + mrow * D_ + mcol);
        async_store_b128(gdst,      l);
        async_store_b128(gdst + 8,  l + 16);
        async_store_b128(gdst + 16, l + 32);
        async_store_b128(gdst + 24, l + 48);
      }
      { __bf16* tmp = h_cur; h_cur = h_nxt; h_nxt = tmp; }
    }
  }
  wait_async0();   // belt-and-braces (S_ENDPGM also waits idle)
}

// ---------------- generic batched GEMM: C = A(MxK) @ W(NxK)^T + bias ----------------
__global__ void gemm_bf16_kernel(const __bf16* __restrict__ A, long sA,
                                 const __bf16* __restrict__ W, long sW,
                                 const float* __restrict__ bias, long sBias,
                                 float* __restrict__ C, long sC,
                                 int M, int N, int K)
{
  const int z = blockIdx.z;
  const __bf16* Ab = A + (long)z * sA;
  const __bf16* Wb = W + (long)z * sW;
  const float*  bb = bias ? bias + (long)z * sBias : nullptr;
  float*        Cb = C + (long)z * sC;

  const int wave = threadIdx.x >> 5;
  const int lane = threadIdx.x & 31;
  const int tilesN = N >> 4;
  const int tile = blockIdx.x * 8 + wave;
  if (tile >= (M >> 4) * tilesN) return;
  const int tm = (tile / tilesN) << 4;
  const int tn = (tile % tilesN) << 4;

  v8f acc;
  const float bi = bb ? bb[tn + (lane & 15)] : 0.f;
  #pragma unroll
  for (int i = 0; i < 8; ++i) acc[i] = bi;

  for (int ks = 0; ks < (K >> 5); ++ks)
    acc = wmma_bf16(ld_tile16(Ab, K, tm, ks * 32, lane),
                    ld_tile16(Wb, K, tn, ks * 32, lane), acc);

  const int mlo = (lane & 16) ? 8 : 0;
  const int n   = tn + (lane & 15);
  #pragma unroll
  for (int r = 0; r < 8; ++r)
    Cb[(long)(tm + r + mlo) * N + n] = acc[r];
}

// ---------------- attention: per (e,b), K/V projected on the fly with WMMA ----------------
__global__ void attn_kernel(const __bf16* __restrict__ enc, const float* __restrict__ qf,
                            const __bf16* __restrict__ Wqkv, const float* __restrict__ bqkv,
                            __bf16* __restrict__ obuf)
{
  const int b = blockIdx.x, e = blockIdx.y;
  const int tid  = threadIdx.x;            // 256 threads = 8 waves
  const int wave = tid >> 5, lane = tid & 31;

  extern __shared__ unsigned char sm[];
  float* q_l  = (float*)sm;                // 4 x 256
  float* sc   = q_l + NF_ * D_;            // 16 x 1024 scores [(h*4+q)][t]
  float* kv   = sc + 16 * N_;              // 16 x 256 projected K or V tile
  float* red  = kv + 16 * D_;              // 16 x 16 reduce scratch
  float* rmax = red + 256;                 // 16
  float* rsum = rmax + 16;                 // 16
  __bf16* et  = (__bf16*)(rsum + 16);      // 16 x 256 enc tile

  const __bf16* Wk = Wqkv + (long)e * (3 * D_) * D_;   // rows [0,256)=Wq [256,512)=Wk [512,768)=Wv
  const float*  bq = bqkv + (long)e * (3 * D_);
  const int str = tid >> 4, stc = (tid & 15) * 16;     // staging row/col

  // load q (dec projection) for this b: q_l[nf][d]
  {
    const int nf = tid >> 6, d = (tid & 63) * 4;
    *(float4*)(q_l + nf * D_ + d) =
        *(const float4*)(qf + (((long)e * NF_ + nf) * B_ + b) * D_ + d);
  }
  __syncthreads();

  // ---- pass 1: scores ----
  for (int tt = 0; tt < N_ / 16; ++tt) {
    antihoist();     // Wk tiles are tt-invariant: keep them streaming from L2
    const int t0 = tt * 16;
    { // stage 16 x 256 enc tile via async DMA
      const __bf16* g = enc + (((long)e * N_ + t0 + str) * B_ + b) * D_ + stc;
      const unsigned l = lds_addr_of(et + str * D_ + stc);
      async_load_b128(l,      g);
      async_load_b128(l + 16, g + 8);
    }
    wait_async0();
    __syncthreads();
    #pragma unroll
    for (int ctl = 0; ctl < 2; ++ctl) {   // K projection: 16 col tiles over 8 waves
      const int ct = wave + ctl * 8;
      v8f acc;
      const float bi = bq[D_ + ct * 16 + (lane & 15)];
      #pragma unroll
      for (int i = 0; i < 8; ++i) acc[i] = bi;
      #pragma unroll
      for (int ks = 0; ks < 8; ++ks)
        acc = wmma_bf16(ld_tile16(et, D_, 0, ks * 32, lane),
                        ld_tile16(Wk, D_, D_ + ct * 16, ks * 32, lane), acc);
      const int mlo = (lane & 16) ? 8 : 0;
      const int n = ct * 16 + (lane & 15);
      #pragma unroll
      for (int r = 0; r < 8; ++r) kv[(r + mlo) * D_ + n] = acc[r];
    }
    __syncthreads();
    { // scores: one (h,q,t) per thread, 64-MAC dot
      const int h = tid >> 6, q = (tid >> 4) & 3, tr = tid & 15;
      float s = 0.f;
      for (int d = 0; d < DH_; ++d)
        s += q_l[q * D_ + h * DH_ + d] * kv[tr * D_ + h * DH_ + d];
      sc[(h * 4 + q) * N_ + t0 + tr] = s * 0.125f;   // 1/sqrt(64)
    }
    __syncthreads();
  }

  // ---- exact softmax over 16 rows of 1024 ----
  const int row = tid >> 4, seg = tid & 15;
  {
    float m = -3.4e38f;
    for (int j = 0; j < 64; ++j) m = fmaxf(m, sc[row * N_ + seg * 64 + j]);
    red[row * 16 + seg] = m;
  }
  __syncthreads();
  if (tid < 16) {
    float m = -3.4e38f;
    for (int s = 0; s < 16; ++s) m = fmaxf(m, red[tid * 16 + s]);
    rmax[tid] = m;
  }
  __syncthreads();
  {
    const float mm = rmax[row];
    float s = 0.f;
    for (int j = 0; j < 64; ++j) {
      const float ex = __expf(sc[row * N_ + seg * 64 + j] - mm);
      sc[row * N_ + seg * 64 + j] = ex;
      s += ex;
    }
    red[row * 16 + seg] = s;
  }
  __syncthreads();
  if (tid < 16) {
    float s = 0.f;
    for (int sg = 0; sg < 16; ++sg) s += red[tid * 16 + sg];
    rsum[tid] = s;
  }
  __syncthreads();
  {
    const float inv = 1.0f / rsum[row];
    for (int j = 0; j < 64; ++j) sc[row * N_ + seg * 64 + j] *= inv;
  }
  __syncthreads();

  // ---- pass 2: out = att @ V, V projected on the fly ----
  float oacc[NF_] = {0.f, 0.f, 0.f, 0.f};   // thread owns (nf=0..3, col=tid)
  const int h2 = tid >> 6;
  for (int tt = 0; tt < N_ / 16; ++tt) {
    antihoist();     // Wv tiles are tt-invariant: keep them streaming from L2
    const int t0 = tt * 16;
    {
      const __bf16* g = enc + (((long)e * N_ + t0 + str) * B_ + b) * D_ + stc;
      const unsigned l = lds_addr_of(et + str * D_ + stc);
      async_load_b128(l,      g);
      async_load_b128(l + 16, g + 8);
    }
    wait_async0();
    __syncthreads();
    #pragma unroll
    for (int ctl = 0; ctl < 2; ++ctl) {   // V projection
      const int ct = wave + ctl * 8;
      v8f acc;
      const float bi = bq[2 * D_ + ct * 16 + (lane & 15)];
      #pragma unroll
      for (int i = 0; i < 8; ++i) acc[i] = bi;
      #pragma unroll
      for (int ks = 0; ks < 8; ++ks)
        acc = wmma_bf16(ld_tile16(et, D_, 0, ks * 32, lane),
                        ld_tile16(Wk, D_, 2 * D_ + ct * 16, ks * 32, lane), acc);
      const int mlo = (lane & 16) ? 8 : 0;
      const int n = ct * 16 + (lane & 15);
      #pragma unroll
      for (int r = 0; r < 8; ++r) kv[(r + mlo) * D_ + n] = acc[r];
    }
    __syncthreads();
    #pragma unroll
    for (int nf = 0; nf < NF_; ++nf)
      for (int tr = 0; tr < 16; ++tr)
        oacc[nf] += sc[(h2 * 4 + nf) * N_ + t0 + tr] * kv[tr * D_ + tid];
    __syncthreads();
  }
  #pragma unroll
  for (int nf = 0; nf < NF_; ++nf)
    obuf[(((long)e * NF_ + nf) * B_ + b) * D_ + tid] = (__bf16)oacc[nf];
}

// ---------------- mean over NF + bf16 cast ----------------
__global__ void poolmean_kernel(const float* __restrict__ oproj, __bf16* __restrict__ pin) {
  const long i = (long)blockIdx.x * blockDim.x + threadIdx.x;   // E*B*D
  const int d = i & 255, b = (i >> 8) & 63, e = (int)(i >> 14);
  float s = 0.f;
  for (int nf = 0; nf < NF_; ++nf) s += oproj[(((long)e * NF_ + nf) * B_ + b) * D_ + d];
  pin[i] = (__bf16)(0.25f * s);
}

// ---------------- LayerNorm ----------------
__global__ void ln_kernel(const float* __restrict__ pooled, const float* __restrict__ lnw,
                          const float* __restrict__ lnb, float* __restrict__ states) {
  const int b = blockIdx.x, e = blockIdx.y, tid = threadIdx.x;
  __shared__ float buf[256];
  const float v = pooled[((long)e * B_ + b) * D_ + tid];
  buf[tid] = v; __syncthreads();
  for (int s = 128; s > 0; s >>= 1) { if (tid < s) buf[tid] += buf[tid + s]; __syncthreads(); }
  const float mu = buf[0] * (1.f / 256.f);
  __syncthreads();
  const float dv = v - mu;
  buf[tid] = dv * dv; __syncthreads();
  for (int s = 128; s > 0; s >>= 1) { if (tid < s) buf[tid] += buf[tid + s]; __syncthreads(); }
  const float var = buf[0] * (1.f / 256.f);
  states[((long)e * B_ + b) * D_ + tid] =
      dv * rsqrtf(var + 1e-5f) * lnw[e * D_ + tid] + lnb[e * D_ + tid];
}

// ---------------- gating, top-2 combine, output assembly ----------------
__global__ void combine_kernel(const float* __restrict__ zs, const float* __restrict__ states,
                               const float* __restrict__ Wgate, const float* __restrict__ bgate,
                               float* __restrict__ out) {
  const int b = blockIdx.x, tid = threadIdx.x;
  __shared__ float sp[4];
  __shared__ float w[4];
  if (tid < 4) {
    float s = bgate[tid];
    for (int d = 0; d < D_; ++d) s += zs[b * D_ + d] * Wgate[tid * D_ + d];
    sp[tid] = s;
  }
  __syncthreads();
  if (tid == 0) {
    const float m = fmaxf(fmaxf(sp[0], sp[1]), fmaxf(sp[2], sp[3]));
    float p[4], sum = 0.f;
    for (int i = 0; i < 4; ++i) { p[i] = __expf(sp[i] - m); sum += p[i]; }
    for (int i = 0; i < 4; ++i) { p[i] /= sum; sp[i] = p[i]; }
    int i0 = 0;
    for (int i = 1; i < 4; ++i) if (p[i] > p[i0]) i0 = i;
    int i1 = -1;
    for (int i = 0; i < 4; ++i) if (i != i0 && (i1 < 0 || p[i] > p[i1])) i1 = i;
    const float inv = 1.0f / (p[i0] + p[i1] + 1e-9f);
    for (int i = 0; i < 4; ++i) w[i] = 0.f;
    w[i0] = p[i0] * inv;
    w[i1] = p[i1] * inv;
  }
  __syncthreads();
  float comb = 0.f;
  for (int s4 = 0; s4 < 4; ++s4)
    comb += w[s4] * states[((long)(4 + s4) * B_ + b) * D_ + tid];
  for (int k = 0; k < 4; ++k)
    out[((long)k * B_ + b) * D_ + tid] = states[((long)k * B_ + b) * D_ + tid];
  out[((long)4 * B_ + b) * D_ + tid] = comb;
  for (int k = 0; k < 4; ++k)
    out[((long)(5 + k) * B_ + b) * D_ + tid] = states[((long)(4 + k) * B_ + b) * D_ + tid];
  if (tid < 4) out[(long)9 * B_ * D_ + b * 4 + tid] = sp[tid];
}

// ---------------- host orchestration ----------------
extern "C" void kernel_launch(void* const* d_in, const int* in_sizes, int n_in,
                              void* d_out, int out_size, void* d_ws, size_t ws_size,
                              hipStream_t stream) {
  (void)in_sizes; (void)n_in; (void)out_size; (void)ws_size;
  const float* Z       = (const float*)d_in[0];
  const float* c_h     = (const float*)d_in[1];
  const float* Wih_enc = (const float*)d_in[2];
  const float* Whh_enc = (const float*)d_in[3];
  const float* b_enc   = (const float*)d_in[4];
  const float* Wih_dec = (const float*)d_in[5];
  const float* Whh_dec = (const float*)d_in[6];
  const float* b_dec   = (const float*)d_in[7];
  const float* Wqkv    = (const float*)d_in[8];
  const float* bqkv    = (const float*)d_in[9];
  const float* Wo      = (const float*)d_in[10];
  const float* bo      = (const float*)d_in[11];
  const float* Wh0     = (const float*)d_in[12];
  const float* bh0     = (const float*)d_in[13];
  const float* Wc0     = (const float*)d_in[14];
  const float* bc0     = (const float*)d_in[15];
  const float* Wpool   = (const float*)d_in[16];
  const float* bpool   = (const float*)d_in[17];
  const float* ln_w    = (const float*)d_in[18];
  const float* ln_b    = (const float*)d_in[19];
  const float* Wproj   = (const float*)d_in[20];
  const float* bproj   = (const float*)d_in[21];
  const float* Wgate   = (const float*)d_in[22];
  const float* bgate   = (const float*)d_in[23];
  float* out = (float*)d_out;

  // workspace carve-out (~332 MB, dominated by bf16 encoder states)
  char* ws = (char*)d_ws;
  auto alloc = [&](size_t bytes) {
    char* p = ws; ws += (bytes + 255) & ~(size_t)255; return p;
  };
  __bf16* Zbf    = (__bf16*)alloc((size_t)B_ * N_ * D_ * 2);
  __bf16* WihEb  = (__bf16*)alloc((size_t)E_ * G4_ * D_ * 2);
  __bf16* WhhEb  = (__bf16*)alloc((size_t)E_ * G4_ * D_ * 2);
  __bf16* WihDb  = (__bf16*)alloc((size_t)E_ * G4_ * D_ * 2);
  __bf16* WhhDb  = (__bf16*)alloc((size_t)E_ * G4_ * D_ * 2);
  __bf16* Wqkvb  = (__bf16*)alloc((size_t)E_ * 3 * D_ * D_ * 2);
  __bf16* Wob    = (__bf16*)alloc((size_t)E_ * D_ * D_ * 2);
  __bf16* Wpoolb = (__bf16*)alloc((size_t)E_ * D_ * D_ * 2);
  __bf16* Wprojb = (__bf16*)alloc((size_t)D_ * D_ * 2);
  __bf16* encb   = (__bf16*)alloc((size_t)E_ * N_ * B_ * D_ * 2);
  __bf16* decb   = (__bf16*)alloc((size_t)E_ * NF_ * B_ * D_ * 2);
  float*  qf     = (float*)alloc((size_t)E_ * NF_ * B_ * D_ * 4);
  __bf16* ob     = (__bf16*)alloc((size_t)E_ * NF_ * B_ * D_ * 2);
  float*  oprojf = (float*)alloc((size_t)E_ * NF_ * B_ * D_ * 4);
  __bf16* pinb   = (__bf16*)alloc((size_t)E_ * B_ * D_ * 2);
  float*  pooledf= (float*)alloc((size_t)E_ * B_ * D_ * 4);
  float*  statesf= (float*)alloc((size_t)E_ * B_ * D_ * 4);
  __bf16* zmb    = (__bf16*)alloc((size_t)B_ * D_ * 2);
  float*  zsf    = (float*)alloc((size_t)B_ * D_ * 4);

  auto cvt = [&](const float* s, __bf16* d, long n) {
    int grid = (int)((n + 1023) / 1024); if (grid > 4096) grid = 4096; if (grid < 1) grid = 1;
    cvt_kernel<<<grid, 256, 0, stream>>>(s, d, n);
  };
  cvt(Z,       Zbf,    (long)B_ * N_ * D_);
  cvt(Wih_enc, WihEb,  (long)E_ * G4_ * D_);
  cvt(Whh_enc, WhhEb,  (long)E_ * G4_ * D_);
  cvt(Wih_dec, WihDb,  (long)E_ * G4_ * D_);
  cvt(Whh_dec, WhhDb,  (long)E_ * G4_ * D_);
  cvt(Wqkv,    Wqkvb,  (long)E_ * 3 * D_ * D_);
  cvt(Wo,      Wob,    (long)E_ * D_ * D_);
  cvt(Wpool,   Wpoolb, (long)E_ * D_ * D_);
  cvt(Wproj,   Wprojb, (long)D_ * D_);
  zmean_kernel<<<B_, D_, 0, stream>>>(Z, zmb);

  // fused encoder+decoder scan: 1 block / expert, 160 KB dynamic LDS
  const size_t scan_lds = (size_t)3 * B_ * D_ * sizeof(__bf16)   // x + h double buffer
                        + (size_t)B_ * D_ * sizeof(float);       // c state
  lstm_scan_kernel<<<E_, 512, scan_lds, stream>>>(
      Zbf, c_h, WihEb, WhhEb, b_enc, WihDb, WhhDb, b_dec,
      Wh0, bh0, Wc0, bc0, encb, decb);

  // q = dec @ Wq^T + bq   (M=NF*B=256, N=256, K=256, batched over experts)
  gemm_bf16_kernel<<<dim3(32, 1, E_), 256, 0, stream>>>(
      decb, (long)NF_ * B_ * D_, Wqkvb, (long)3 * D_ * D_, bqkv, (long)3 * D_,
      qf, (long)NF_ * B_ * D_, NF_ * B_, D_, D_);

  // attention with on-the-fly K/V projection
  const size_t attn_lds = (size_t)(NF_ * D_ + 16 * N_ + 16 * D_ + 256 + 32) * 4
                        + (size_t)16 * D_ * 2;
  attn_kernel<<<dim3(B_, E_), 256, attn_lds, stream>>>(encb, qf, Wqkvb, bqkv, ob);

  // o @ Wo^T + bo
  gemm_bf16_kernel<<<dim3(32, 1, E_), 256, 0, stream>>>(
      ob, (long)NF_ * B_ * D_, Wob, (long)D_ * D_, bo, (long)D_,
      oprojf, (long)NF_ * B_ * D_, NF_ * B_, D_, D_);

  // mean over NF, then @ Wpool^T + bpool
  poolmean_kernel<<<(E_ * B_ * D_) / 256, 256, 0, stream>>>(oprojf, pinb);
  gemm_bf16_kernel<<<dim3(8, 1, E_), 256, 0, stream>>>(
      pinb, (long)B_ * D_, Wpoolb, (long)D_ * D_, bpool, (long)D_,
      pooledf, (long)B_ * D_, B_, D_, D_);

  ln_kernel<<<dim3(B_, E_), 256, 0, stream>>>(pooledf, ln_w, ln_b, statesf);

  // zs = mean_t(Z) @ Wproj^T + bproj
  gemm_bf16_kernel<<<dim3(8, 1, 1), 256, 0, stream>>>(
      zmb, 0, Wprojb, 0, bproj, 0, zsf, 0, B_, D_, D_);

  combine_kernel<<<B_, 256, 0, stream>>>(zsf, statesf, Wgate, bgate, out);
}
